// GRUModel1_43404939493530
// MI455X (gfx1250) — matrix operand order repaired and checked by reading, not compiled, not run on hardware
//
#include <hip/hip_runtime.h>
#include <hip/hip_bf16.h>
#include <math.h>

#define B_   32
#define T_   1024
#define F_   128
#define D_   64
#define NCLS 11
#define EPS_ 1e-3f
#define KT_  32          // keys per attention iteration

typedef _Float16 half_t;
typedef __attribute__((ext_vector_type(16))) _Float16 v16h;
typedef __attribute__((ext_vector_type(8)))  _Float16 v8h;
typedef __attribute__((ext_vector_type(8)))  float    v8f;
typedef int v4i __attribute__((vector_size(16)));
typedef __attribute__((address_space(1))) v4i* gv4i_p;
typedef __attribute__((address_space(3))) v4i* lv4i_p;

__device__ __forceinline__ v8f wmma_f16(v16h a, v16h b, v8f c) {
  // 8 args: (neg_a, A, neg_b, B, c_mod, C, reuse_a, reuse_b)
  return __builtin_amdgcn_wmma_f32_16x16x32_f16(false, a, false, b, (short)0, c, false, false);
}

// Row reduction across the 16-lane half of a wave32 (C-matrix rows span 16 lanes).
__device__ __forceinline__ float half_rmax(float v) {
  v = fmaxf(v, __shfl_xor(v, 1, 32));
  v = fmaxf(v, __shfl_xor(v, 2, 32));
  v = fmaxf(v, __shfl_xor(v, 4, 32));
  v = fmaxf(v, __shfl_xor(v, 8, 32));
  return v;
}
__device__ __forceinline__ float half_rsum(float v) {
  v += __shfl_xor(v, 1, 32);
  v += __shfl_xor(v, 2, 32);
  v += __shfl_xor(v, 4, 32);
  v += __shfl_xor(v, 8, 32);
  return v;
}

// 16-byte global -> LDS copy. Uses the gfx1250 async-to-LDS path (ASYNCcnt)
// when the toolchain exposes it; falls back to a synchronous staged copy
// (same barrier protocol) otherwise.
#if defined(__has_builtin)
#if __has_builtin(__builtin_amdgcn_global_load_async_to_lds_b128)
#define HAVE_ASYNC_LDS 1
#endif
#endif

__device__ __forceinline__ void copy16_g2l(const half_t* g, half_t* l) {
#ifdef HAVE_ASYNC_LDS
  void* gv = (void*)g;   // drop const; C-style addrspace casts below
  void* lv = (void*)l;
  __builtin_amdgcn_global_load_async_to_lds_b128((gv4i_p)gv, (lv4i_p)lv, 0, 0);
#else
  *(v8h*)l = *(const v8h*)g;
#endif
}

__device__ __forceinline__ void wait_async_le4() {
#ifdef HAVE_ASYNC_LDS
  asm volatile("s_wait_asynccnt 0x4" ::: "memory");
#endif
}
__device__ __forceinline__ void wait_async_0() {
#ifdef HAVE_ASYNC_LDS
  asm volatile("s_wait_asynccnt 0x0" ::: "memory");
#endif
}

// ---------------------------------------------------------------------------
// Kernel 1: xp = x + sinusoidal positional encoding, stored f16.
// ---------------------------------------------------------------------------
__global__ void k_pe(const float* __restrict__ x, half_t* __restrict__ xp, int n) {
  int i = blockIdx.x * blockDim.x + threadIdx.x;
  if (i >= n) return;
  int f = i & (F_ - 1);
  int t = (i / F_) & (T_ - 1);
  float fe  = (float)(f & ~1);
  float div = __expf(-9.210340371976184f * fe / (float)F_);  // ln(10000)
  float ang = (float)t * div;
  float pe  = (f & 1) ? cosf(ang) : sinf(ang);
  xp[i] = (half_t)(x[i] + pe);
}

// ---------------------------------------------------------------------------
// Kernel 2: Q/K/V projections via WMMA. Q,K stored [b*T+t][d] f16.
// V stored transposed: vT[b][d][t] f16 so PV B-fragments are contiguous.
// grid = (B*T/16, 3), block = 128 (4 waves; wave w owns N-tile w of D=64).
// ---------------------------------------------------------------------------
__global__ void __launch_bounds__(128) k_proj(
    const half_t* __restrict__ xp,
    const float* __restrict__ wq, const float* __restrict__ wk, const float* __restrict__ wv,
    half_t* __restrict__ q, half_t* __restrict__ k, half_t* __restrict__ vT) {
  int lane = threadIdx.x & 31;
  int nt   = threadIdx.x >> 5;       // which 16-col tile of D
  int rt   = blockIdx.x;             // 16-row tile of B*T
  int mat  = blockIdx.y;             // 0=q 1=k 2=v
  const float* w = (mat == 0) ? wq : (mat == 1) ? wk : wv;
  int g0 = rt * 16;

  int m    = lane & 15;
  int kb8  = (lane >> 4) * 8;
  int kb16 = (lane >> 4) * 16;
  int n    = lane & 15;

  v8f c = {};
  #pragma unroll
  for (int ks = 0; ks < 4; ++ks) {
    // A fragment: xp rows g0.., cols ks*32.. (16x32 f16, contiguous k = 16h + s)
    const half_t* ap = xp + (size_t)(g0 + m) * F_ + ks * 32 + kb16;
    v16h a;
    *(v8h*)&a       = *(const v8h*)(ap);
    *((v8h*)&a + 1) = *(const v8h*)(ap + 8);
    // B fragment: w[(ks*32 + kb16 + j)*64 + nt*16 + n]  (32x16)
    v16h b;
    #pragma unroll
    for (int j = 0; j < 16; ++j)
      b[j] = (half_t)w[(size_t)(ks * 32 + kb16 + j) * D_ + nt * 16 + n];
    c = wmma_f16(a, b, c);
  }

  int col = nt * 16 + n;
  if (mat < 2) {
    half_t* dst = (mat == 0) ? q : k;
    #pragma unroll
    for (int r = 0; r < 8; ++r) {
      int ro = (lane < 16) ? r : r + 8;
      dst[(size_t)(g0 + ro) * D_ + col] = (half_t)c[r];
    }
  } else {
    int bb = g0 / T_;
    int t0 = g0 & (T_ - 1);
    #pragma unroll
    for (int r = 0; r < 8; ++r) {
      int ro = (lane < 16) ? r : r + 8;
      vT[((size_t)bb * D_ + col) * T_ + (t0 + ro)] = (half_t)c[r];
    }
  }
}

// ---------------------------------------------------------------------------
// Kernel 3: flash attention, 4 waves per block covering 64 query rows.
// K/V tiles (32 keys) are staged once per block into LDS with async
// global->LDS copies (ASYNCcnt), double buffered; each wave runs online
// softmax on its own 16x32 score tile. BN-a fused into the epilogue.
// grid = (T/64, B), block = 128.
// ---------------------------------------------------------------------------
__global__ void __launch_bounds__(128) k_attn(
    const half_t* __restrict__ q, const half_t* __restrict__ kmat,
    const half_t* __restrict__ vT,
    const float* __restrict__ gam, const float* __restrict__ bet,
    const float* __restrict__ mu,  const float* __restrict__ var,
    float* __restrict__ x1) {
  __shared__ alignas(32) half_t Ks[2][KT_ * D_];   // [key][d], 4 KB each
  __shared__ alignas(32) half_t Vs[2][D_ * KT_];   // [d][key], 4 KB each
  __shared__ alignas(32) half_t Pl[4][16 * 32];    // per-wave P staging

  int tid  = threadIdx.x;
  int lane = tid & 31;
  int wv   = tid >> 5;
  int qt   = blockIdx.x;
  int bb   = blockIdx.y;
  int g0   = bb * T_ + qt * 64 + wv * 16;

  int mm   = lane & 15;
  int kb8  = (lane >> 4) * 8;
  int kb16 = (lane >> 4) * 16;
  int nn   = lane & 15;

  const half_t* krow0 = kmat + (size_t)bb * T_ * D_;
  const half_t* vrow0 = vT   + (size_t)bb * D_ * T_;

  // Q A-fragments (per wave): d=0..31 and d=32..63
  v16h qa0, qa1;
  {
    const half_t* ap = q + (size_t)(g0 + mm) * D_ + kb16;
    *(v8h*)&qa0       = *(const v8h*)(ap);
    *((v8h*)&qa0 + 1) = *(const v8h*)(ap + 8);
    ap += 32;
    *(v8h*)&qa1       = *(const v8h*)(ap);
    *((v8h*)&qa1 + 1) = *(const v8h*)(ap + 8);
  }

  // Stage one 32-key tile (K: 4KB, V^T: 4KB) — 2+2 16B chunks per thread.
  auto stage = [&](int buf, int kt0) {
    #pragma unroll
    for (int c = tid; c < 256; c += 128)                     // K tile: contiguous
      copy16_g2l(krow0 + (size_t)kt0 * D_ + c * 8, &Ks[buf][c * 8]);
    #pragma unroll
    for (int c = tid; c < 256; c += 128) {                   // V^T tile: 64 rows x 32
      int d  = c >> 2;
      int xo = (c & 3) * 8;
      copy16_g2l(vrow0 + (size_t)d * T_ + kt0 + xo, &Vs[buf][d * KT_ + xo]);
    }
  };

  float mrow[8], lrow[8];
  #pragma unroll
  for (int r = 0; r < 8; ++r) { mrow[r] = -INFINITY; lrow[r] = 0.f; }
  v8f o0 = {}, o1 = {}, o2 = {}, o3 = {};

  stage(0, 0);

  for (int it = 0; it < T_ / KT_; ++it) {
    int buf = it & 1;
    if (it + 1 < T_ / KT_) {
      stage(buf ^ 1, (it + 1) * KT_);   // prefetch next tile (other buffer)
      wait_async_le4();                  // our 4 chunks of current tile are done
    } else {
      wait_async_0();
    }
    __syncthreads();                     // all waves' chunks of `buf` landed

    // ---- S = Q K^T for 32 keys (two 16x16 N-tiles, K=64 in two steps) ----
    v8f s0 = {}, s1 = {};
    {
      const half_t* kr = &Ks[buf][(size_t)nn * D_ + kb16];
      v16h b0 = *(const v16h*)(kr);
      v16h b1 = *(const v16h*)(kr + 32);
      s0 = wmma_f16(qa0, b0, s0);
      s0 = wmma_f16(qa1, b1, s0);
      const half_t* kr2 = &Ks[buf][(size_t)(16 + nn) * D_ + kb16];
      v16h b2 = *(const v16h*)(kr2);
      v16h b3 = *(const v16h*)(kr2 + 32);
      s1 = wmma_f16(qa0, b2, s1);
      s1 = wmma_f16(qa1, b3, s1);
    }
    // ---- online softmax update ----
    #pragma unroll
    for (int r = 0; r < 8; ++r) {
      float a0 = s0[r] * 0.125f;                 // 1/sqrt(D)
      float a1 = s1[r] * 0.125f;
      float rm = half_rmax(fmaxf(a0, a1));
      float nm = fmaxf(mrow[r], rm);
      float alpha = __expf(mrow[r] - nm);        // first iter: exp(-inf)=0
      float p0 = __expf(a0 - nm);
      float p1 = __expf(a1 - nm);
      lrow[r] = lrow[r] * alpha + half_rsum(p0 + p1);
      mrow[r] = nm;
      o0[r] *= alpha; o1[r] *= alpha; o2[r] *= alpha; o3[r] *= alpha;
      int ro = (lane < 16) ? r : r + 8;
      Pl[wv][ro * 32 + nn]      = (half_t)p0;
      Pl[wv][ro * 32 + nn + 16] = (half_t)p1;
    }
    __syncthreads();                     // make cross-lane P staging visible
    // ---- P A-fragment from LDS (16x32) ----
    v16h pa;
    {
      const half_t* pp = &Pl[wv][mm * 32 + kb16];
      *(v8h*)&pa       = *(const v8h*)(pp);
      *((v8h*)&pa + 1) = *(const v8h*)(pp + 8);
    }
    // ---- O += P V ----
    {
      const half_t* vr = &Vs[buf][(size_t)nn * KT_ + kb16];
      v16h vb0 = *(const v16h*)(vr);
      v16h vb1 = *(const v16h*)(vr + 16 * KT_);
      v16h vb2 = *(const v16h*)(vr + 32 * KT_);
      v16h vb3 = *(const v16h*)(vr + 48 * KT_);
      o0 = wmma_f16(pa, vb0, o0);
      o1 = wmma_f16(pa, vb1, o1);
      o2 = wmma_f16(pa, vb2, o2);
      o3 = wmma_f16(pa, vb3, o3);
    }
    __syncthreads();                     // all reads of `buf` done before reuse
  }

  // ---- epilogue: normalize + BN-a, store x1 f32 ----
  #pragma unroll
  for (int r = 0; r < 8; ++r) {
    float inv = 1.0f / lrow[r];
    int ro = (lane < 16) ? r : r + 8;
    float* dst = x1 + (size_t)(g0 + ro) * D_;
    float vals[4] = { o0[r], o1[r], o2[r], o3[r] };
    #pragma unroll
    for (int dt = 0; dt < 4; ++dt) {
      int d = dt * 16 + nn;
      float v = vals[dt] * inv;
      v = (v - mu[d]) * (gam[d] * __frsqrt_rn(var[d] + EPS_)) + bet[d];
      dst[d] = v;
    }
  }
}

// ---------------------------------------------------------------------------
// Kernel 4: per-row head. Weights staged in LDS (~56 KB << 320 KB/WGP).
// Each thread owns one (b,t) row. grid = B*T/256, block = 256.
// ---------------------------------------------------------------------------
__global__ void __launch_bounds__(256) k_head(
    const float* __restrict__ x,  const float* __restrict__ x1,
    const float* __restrict__ w0, const float* __restrict__ b0,
    const float* __restrict__ w1, const float* __restrict__ b1,
    const float* __restrict__ g1, const float* __restrict__ be1,
    const float* __restrict__ mu1, const float* __restrict__ var1,
    const float* __restrict__ wd, const float* __restrict__ bd,
    float* __restrict__ out) {
  __shared__ float w0s[D_ * 64];
  __shared__ float w1s[F_ * 64];
  __shared__ float wds[128 * NCLS];
  __shared__ float b0s[64], b1s[64], bds[NCLS];
  __shared__ float scl[128], sft[128];
  int tid = threadIdx.x;
  for (int i = tid; i < D_ * 64; i += 256) w0s[i] = w0[i];
  for (int i = tid; i < F_ * 64; i += 256) w1s[i] = w1[i];
  for (int i = tid; i < 128 * NCLS; i += 256) wds[i] = wd[i];
  if (tid < 64) { b0s[tid] = b0[tid]; b1s[tid] = b1[tid]; }
  if (tid < NCLS) bds[tid] = bd[tid];
  if (tid < 128) {
    float s = g1[tid] * __frsqrt_rn(var1[tid] + EPS_);
    scl[tid] = s;
    sft[tid] = be1[tid] - mu1[tid] * s;
  }
  __syncthreads();

  int g = blockIdx.x * 256 + tid;          // 0..32767
  const float* xr  = x  + (size_t)g * F_;
  const float* x1r = x1 + (size_t)g * D_;

  float h[128];
  for (int e = 0; e < 64; ++e) {
    float acc = b0s[e];
    #pragma unroll 8
    for (int d = 0; d < D_; ++d) acc += x1r[d] * w0s[d * 64 + e];
    h[e] = fmaxf(acc, 0.f);
  }
  for (int e = 0; e < 64; ++e) {
    float acc = b1s[e];
    #pragma unroll 8
    for (int f = 0; f < F_; ++f) acc += xr[f] * w1s[f * 64 + e];
    h[64 + e] = fmaxf(acc, 0.f);
  }
  for (int i = 0; i < 128; ++i) h[i] = h[i] * scl[i] + sft[i];

  float lg[NCLS];
  float mx = -INFINITY;
  for (int c = 0; c < NCLS; ++c) {
    float acc = bds[c];
    #pragma unroll 8
    for (int e = 0; e < 128; ++e) acc += h[e] * wds[e * NCLS + c];
    lg[c] = acc;
    mx = fmaxf(mx, acc);
  }
  float s = 0.f;
  for (int c = 0; c < NCLS; ++c) { lg[c] = __expf(lg[c] - mx); s += lg[c]; }
  float invs = 1.f / s;
  float* o = out + (size_t)g * NCLS;
  for (int c = 0; c < NCLS; ++c) o[c] = lg[c] * invs;
}

// ---------------------------------------------------------------------------
extern "C" void kernel_launch(void* const* d_in, const int* in_sizes, int n_in,
                              void* d_out, int out_size, void* d_ws, size_t ws_size,
                              hipStream_t stream) {
  const float* x   = (const float*)d_in[0];
  const float* wq  = (const float*)d_in[1];
  const float* wk  = (const float*)d_in[2];
  const float* wv  = (const float*)d_in[3];
  const float* bag = (const float*)d_in[4];
  const float* bab = (const float*)d_in[5];
  const float* bam = (const float*)d_in[6];
  const float* bav = (const float*)d_in[7];
  const float* w0  = (const float*)d_in[8];
  const float* b0  = (const float*)d_in[9];
  const float* w1  = (const float*)d_in[10];
  const float* b1  = (const float*)d_in[11];
  const float* g1  = (const float*)d_in[12];
  const float* be1 = (const float*)d_in[13];
  const float* mu1 = (const float*)d_in[14];
  const float* vr1 = (const float*)d_in[15];
  const float* wd  = (const float*)d_in[16];
  const float* bd  = (const float*)d_in[17];
  float* out = (float*)d_out;

  char* ws = (char*)d_ws;
  half_t* xp = (half_t*)(ws);                               //  8 MB: [B*T][F] f16
  half_t* qh = (half_t*)(ws + (size_t)8  * 1024 * 1024);    //  4 MB: [B*T][D] f16
  half_t* kh = (half_t*)(ws + (size_t)12 * 1024 * 1024);    //  4 MB: [B*T][D] f16
  half_t* vT = (half_t*)(ws + (size_t)16 * 1024 * 1024);    //  4 MB: [B][D][T] f16
  float*  x1 = (float*) (ws + (size_t)20 * 1024 * 1024);    //  8 MB: [B*T][D] f32

  int npe = B_ * T_ * F_;
  k_pe  <<<dim3((npe + 255) / 256), dim3(256), 0, stream>>>(x, xp, npe);
  k_proj<<<dim3(B_ * T_ / 16, 3),   dim3(128), 0, stream>>>(xp, wq, wk, wv, qh, kh, vT);
  k_attn<<<dim3(T_ / 64, B_),       dim3(128), 0, stream>>>(qh, kh, vT, bag, bab, bam, bav, x1);
  k_head<<<dim3(B_ * T_ / 256),     dim3(256), 0, stream>>>(x, x1, w0, b0, w1, b1,
                                                            g1, be1, mu1, vr1, wd, bd, out);
}
